// GFM_8615704396194
// MI455X (gfx1250) — matrix operand order repaired and checked by reading, not compiled, not run on hardware
//
#include <hip/hip_runtime.h>

typedef __attribute__((ext_vector_type(16))) _Float16 v16h;
typedef __attribute__((ext_vector_type(8)))  float    v8f;
typedef __attribute__((ext_vector_type(4)))  int      v4i;

#define C_DIM 64
#define PTS_PER_BLOCK 64
#define THREADS 128
#define LROW 68   // padded LDS row stride (floats) to avoid bank conflicts

// Async global->LDS path (gfx1250): guarded so absence degrades to the
// proven VGPR-staged path.
#if defined(__has_builtin)
#  if __has_builtin(__builtin_amdgcn_global_load_async_to_lds_b128)
#    define GFM_ASYNC_LDS 1
#  endif
#endif
#ifndef GFM_ASYNC_LDS
#  define GFM_ASYNC_LDS 0
#endif

#if GFM_ASYNC_LDS
typedef __attribute__((address_space(1))) v4i gv4i;   // global b128 element
typedef __attribute__((address_space(3))) v4i lv4i;   // LDS b128 element
#endif

// ---------------------------------------------------------------------------
// Kernel A: zero the v_new accumulator (in d_out) and the counts (in d_ws).
// ---------------------------------------------------------------------------
__global__ void gfm_zero_kernel(float* __restrict__ a, long long na,
                                float* __restrict__ b, long long nb) {
    long long i = (long long)blockIdx.x * blockDim.x + threadIdx.x;
    long long stride = (long long)gridDim.x * blockDim.x;
    for (long long k = i; k < na; k += stride) a[k] = 0.0f;
    for (long long k = i; k < nb; k += stride) b[k] = 0.0f;
}

// ---------------------------------------------------------------------------
// Kernel B: gather voxel rows, WMMA gate logits, softmax, fuse, scatter-add.
// One block = 64 points, 128 threads = 4 waves, 16 points per wave (one
// 16x16 WMMA tile per wave, K=128 = [p|v2p] concat, via 4x wmma_f32_16x16x32_f16).
// ---------------------------------------------------------------------------
__global__ void __launch_bounds__(THREADS)
gfm_fuse_scatter_kernel(const float* __restrict__ p_feats,
                        const float* __restrict__ v_feats,
                        const float* __restrict__ Wp,
                        const float* __restrict__ bp,
                        const float* __restrict__ Wv,
                        const float* __restrict__ bv,
                        const int*   __restrict__ p2v,
                        float* __restrict__ fuse_out,
                        float* __restrict__ vnew_sums,
                        float* __restrict__ counts,
                        int n_pts) {
    __shared__ float lds_p[PTS_PER_BLOCK * LROW];
    __shared__ float lds_v[PTS_PER_BLOCK * LROW];
    __shared__ float lds_logit[PTS_PER_BLOCK * 2];
    __shared__ float lds_w[PTS_PER_BLOCK * 2];
    __shared__ int   lds_idx[PTS_PER_BLOCK];

    const int t  = threadIdx.x;
    const int p0 = blockIdx.x * PTS_PER_BLOCK;
    const bool full = (p0 + PTS_PER_BLOCK) <= n_pts;   // block-uniform

    // ---- stage voxel indices for this tile ----
    if (t < PTS_PER_BLOCK) {
        int gp = p0 + t;
        lds_idx[t] = (gp < n_pts) ? p2v[gp] : 0;
    }
    __syncthreads();

    // ---- phase 1: stage p-tile and gathered v-tile into LDS -------------
    const float4* p4 = (const float4*)p_feats;
    const float4* v4 = (const float4*)v_feats;
    if (full) {
        // hot path: branch-free staging of 2 x 16KB tiles
#if GFM_ASYNC_LDS
#pragma unroll
        for (int i = 0; i < 8; ++i) {
            int e  = i * THREADS + t;        // 0..1023 float4 slots
            int pt = e >> 4;                 // point within tile
            int c4 = e & 15;                 // float4 chunk within 64-ch row
            __builtin_amdgcn_global_load_async_to_lds_b128(
                (gv4i*)(p4 + ((long long)(p0 + pt) * 16 + c4)),
                (lv4i*)((float4*)&lds_p[pt * LROW] + c4), 0, 0);
            __builtin_amdgcn_global_load_async_to_lds_b128(
                (gv4i*)(v4 + ((long long)lds_idx[pt] * 16 + c4)),
                (lv4i*)((float4*)&lds_v[pt * LROW] + c4), 0, 0);
        }
#else
#pragma unroll
        for (int i = 0; i < 8; ++i) {
            int e  = i * THREADS + t;
            int pt = e >> 4;
            int c4 = e & 15;
            ((float4*)&lds_p[pt * LROW])[c4] = p4[(long long)(p0 + pt) * 16 + c4];
            ((float4*)&lds_v[pt * LROW])[c4] = v4[(long long)lds_idx[pt] * 16 + c4];
        }
#endif
    } else {
        // tail block: guarded loads, zero-fill out-of-range rows
#pragma unroll
        for (int i = 0; i < 8; ++i) {
            int e  = i * THREADS + t;
            int pt = e >> 4;
            int c4 = e & 15;
            int gp = p0 + pt;
            float4 pv = make_float4(0.f, 0.f, 0.f, 0.f);
            float4 vv = make_float4(0.f, 0.f, 0.f, 0.f);
            if (gp < n_pts) {
                pv = p4[(long long)gp * 16 + c4];
                vv = v4[(long long)lds_idx[pt] * 16 + c4];
            }
            ((float4*)&lds_p[pt * LROW])[c4] = pv;
            ((float4*)&lds_v[pt * LROW])[c4] = vv;
        }
    }
#if GFM_ASYNC_LDS
#  if defined(__has_builtin) && __has_builtin(__builtin_amdgcn_s_wait_asynccnt)
    __builtin_amdgcn_s_wait_asynccnt(0);
#  else
    asm volatile("s_wait_asynccnt 0" ::: "memory");
#  endif
#endif
    __syncthreads();

    // ---- phase 2: WMMA gate logits -------------------------------------
    const int lane = t & 31;
    const int wave = t >> 5;
    const int col  = lane & 15;   // B/D column (N)
    const int grp  = lane >> 4;   // K half-group select

    // B fragments: combined [Wp;Wv] (128x2) zero-padded to 128x16.
    // B 32x16 layout: lane holds B[Kbase + grp*8 + j][col] (j<8), B[Kbase + 16 + grp*8 + j-8][col] (j>=8).
    v16h bfrag[4];
#pragma unroll
    for (int kb = 0; kb < 4; ++kb) {
        v16h b = {};
        if (col < 2) {
#pragma unroll
            for (int j = 0; j < 16; ++j) {
                int K = kb * 32 + (j < 8 ? grp * 8 + j : 16 + grp * 8 + (j - 8));        // 0..127
                float w = (K < C_DIM) ? Wp[K * 2 + col]
                                      : Wv[(K - C_DIM) * 2 + col];
                b[j] = (_Float16)w;
            }
        }
        bfrag[kb] = b;
    }

    // A fragments: row M = lane&15 of this wave's 16-point tile.
    // 16-bit A 16x32 layout: halves 0..7 = K(kbase+grp*8 ..+7),
    //                        halves 8..15 = K(kbase+16+grp*8 ..+7).
    const float* prow = &lds_p[(wave * 16 + (lane & 15)) * LROW];
    const float* vrow = &lds_v[(wave * 16 + (lane & 15)) * LROW];
    v8f acc = {};
#pragma unroll
    for (int kb = 0; kb < 4; ++kb) {
        const float* src = (kb < 2) ? prow : vrow;   // K<64 -> p, else v2p
        int kk = (kb & 1) * 32;
        v16h a;
#pragma unroll
        for (int j = 0; j < 8; ++j) {
            a[j]     = (_Float16)src[kk + grp * 8 + j];
            a[j + 8] = (_Float16)src[kk + 16 + grp * 8 + j];
        }
        acc = __builtin_amdgcn_wmma_f32_16x16x32_f16(
            /*neg_a=*/false, a, /*neg_b=*/false, bfrag[kb],
            /*c_mod=*/(short)0, acc, /*reuse_a=*/false, /*reuse_b=*/false);
    }

    // C/D layout: lane holds D[M = r + 8*grp][N = col] in acc[r].
    if (col < 2) {
        float bias = bp[col] + bv[col];
#pragma unroll
        for (int r = 0; r < 8; ++r) {
            int pt = wave * 16 + r + 8 * grp;
            lds_logit[pt * 2 + col] = acc[r] + bias;
        }
    }
    __syncthreads();

    // ---- softmax weights + per-point count scatter ----------------------
    if (t < PTS_PER_BLOCK) {
        float l0 = lds_logit[t * 2 + 0];
        float l1 = lds_logit[t * 2 + 1];
        float m  = fmaxf(l0, l1);
        float e0 = __expf(l0 - m);
        float e1 = __expf(l1 - m);
        float inv = 1.0f / (e0 + e1);
        lds_w[t * 2 + 0] = e0 * inv;
        lds_w[t * 2 + 1] = e1 * inv;
        if (p0 + t < n_pts) atomicAdd(&counts[lds_idx[t]], 1.0f);
    }
    __syncthreads();

    // ---- phase 3: fuse, store, scatter-add (L2-resident f32 atomics) ----
#pragma unroll
    for (int i = 0; i < 8; ++i) {
        int e  = i * THREADS + t;
        int pt = e >> 4;
        int c4 = e & 15;
        int gp = p0 + pt;
        if (!full && gp >= n_pts) continue;   // uniform-false on full tiles
        float w0 = lds_w[pt * 2 + 0];
        float w1 = lds_w[pt * 2 + 1];
        float4 pv = ((const float4*)&lds_p[pt * LROW])[c4];
        float4 vv = ((const float4*)&lds_v[pt * LROW])[c4];
        float4 f;
        f.x = pv.x * w0 + vv.x * w1;
        f.y = pv.y * w0 + vv.y * w1;
        f.z = pv.z * w0 + vv.z * w1;
        f.w = pv.w * w0 + vv.w * w1;
        ((float4*)fuse_out)[(long long)gp * 16 + c4] = f;
        float* dst = &vnew_sums[(long long)lds_idx[pt] * C_DIM + c4 * 4];
        atomicAdd(dst + 0, f.x);
        atomicAdd(dst + 1, f.y);
        atomicAdd(dst + 2, f.z);
        atomicAdd(dst + 3, f.w);
    }
}

// ---------------------------------------------------------------------------
// Kernel C: v_new = sums / max(counts, 1)
// ---------------------------------------------------------------------------
__global__ void gfm_finalize_kernel(float* __restrict__ vnew,
                                    const float* __restrict__ counts,
                                    long long n_elems) {
    long long i = (long long)blockIdx.x * blockDim.x + threadIdx.x;
    long long stride = (long long)gridDim.x * blockDim.x;
    for (long long k = i; k < n_elems; k += stride) {
        float cnt = counts[k >> 6];   // C_DIM == 64
        vnew[k] = vnew[k] / fmaxf(cnt, 1.0f);
    }
}

extern "C" void kernel_launch(void* const* d_in, const int* in_sizes, int n_in,
                              void* d_out, int out_size, void* d_ws, size_t ws_size,
                              hipStream_t stream) {
    const float* p_feats = (const float*)d_in[0];
    const float* v_feats = (const float*)d_in[1];
    const float* Wp      = (const float*)d_in[2];
    const float* bp      = (const float*)d_in[3];
    const float* Wv      = (const float*)d_in[4];
    const float* bv      = (const float*)d_in[5];
    const int*   p2v     = (const int*)d_in[6];

    const int n_pts = in_sizes[0] / C_DIM;
    const int n_vox = in_sizes[1] / C_DIM;

    float* fuse_out = (float*)d_out;
    float* vnew     = fuse_out + (long long)n_pts * C_DIM;  // sums -> v_new in place
    float* counts   = (float*)d_ws;                         // n_vox floats

    const long long nv_elems = (long long)n_vox * C_DIM;

    gfm_zero_kernel<<<2048, 256, 0, stream>>>(vnew, nv_elems, counts, (long long)n_vox);

    const int nblocks = (n_pts + PTS_PER_BLOCK - 1) / PTS_PER_BLOCK;
    gfm_fuse_scatter_kernel<<<nblocks, THREADS, 0, stream>>>(
        p_feats, v_feats, Wp, bp, Wv, bv, p2v,
        fuse_out, vnew, counts, n_pts);

    gfm_finalize_kernel<<<4096, 256, 0, stream>>>(vnew, counts, nv_elems);
}
